// DistributionalQNetwork_17987323035731
// MI455X (gfx1250) — matrix operand order repaired and checked by reading, not compiled, not run on hardware
//
#include <hip/hip_runtime.h>

#define V_MIN_C  (-10.0f)
#define V_MAX_C  (10.0f)
#define NATOMS   251
#define BM       128
#define STRH     528          // f16 LDS row stride (512 + 16 pad), keeps 16B alignment
#define NTHREADS 256

typedef __attribute__((ext_vector_type(16))) _Float16 v16h;
typedef __attribute__((ext_vector_type(8)))  _Float16 v8h;
typedef __attribute__((ext_vector_type(8)))  float    v8f;

// ---------------------------------------------------------------------------
// Prepack: W[fi,fo] f32 row-major -> f16 tiles in exact WMMA B-fragment order.
// Tile t = ntile*KC + kc covers K rows [kc*32, kc*32+32) and cols
// [ntile*16, ntile*16+16). Per-lane layout (wave32, 16-bit B 32x16):
//   lane = n_in + ((k_in & 8) ? 16 : 0), half index h = (k_in & 7) + ((k_in & 16) ? 8 : 0),
// stored contiguously: dst[(t*32 + lane)*16 + h].
// ---------------------------------------------------------------------------
__global__ void prepack_w(const float* __restrict__ W, _Float16* __restrict__ dst,
                          int fi, int fo, int foP) {
  int idx = blockIdx.x * blockDim.x + threadIdx.x;
  int total = fi * foP;
  if (idx >= total) return;
  int h    = idx & 15;
  int lane = (idx >> 4) & 31;
  int t    = idx >> 9;
  int KC   = fi >> 5;            // fi is a multiple of 32 for all layers
  int ntile = t / KC;
  int kc    = t - ntile * KC;
  int n = ntile * 16 + (lane & 15);
  int k = kc * 32 + ((lane & 16) ? 8 : 0) + (h & 7) + ((h & 8) ? 16 : 0);
  float v = (n < fo) ? W[k * fo + n] : 0.0f;
  dst[idx] = (_Float16)v;
}

// A-fragment loader: 16-bit A 16x32 layout (two contiguous 8-half runs per lane)
__device__ __forceinline__ v16h load_a(const _Float16* __restrict__ inBuf,
                                       int mbase, int nloc, int kb, int kc) {
  const _Float16* rowp = inBuf + (mbase + nloc) * STRH + kc * 32;
  v8h lo = *(const v8h*)(rowp + kb);        // K = kb .. kb+7
  v8h hi = *(const v8h*)(rowp + 16 + kb);   // K = 16+kb .. 16+kb+7
  v16h a;
#pragma unroll
  for (int i = 0; i < 8; ++i) { a[i] = lo[i]; a[i + 8] = hi[i]; }
  return a;
}

// ---------------------------------------------------------------------------
// One hidden layer: GEMM (WMMA f16->f32, 2-way N blocking: one A load feeds
// two WMMAs) + bias, raw f16 to LDS, wave-local LayerNorm stats via shfl
// reduction, then block-wide LN + SiLU in place.
// ---------------------------------------------------------------------------
template<int FI, int FO>
__device__ __forceinline__ void layer_fwd(
    const _Float16* __restrict__ inBuf,
    _Float16* __restrict__ outBuf,
    const _Float16* __restrict__ Wp,
    const float* __restrict__ bias,
    const float* __restrict__ g,
    const float* __restrict__ be,
    float* __restrict__ stats,
    int tid)
{
  const int lane  = tid & 31;
  const int wave  = tid >> 5;
  const int mbase = wave * 16;
  const int nloc  = lane & 15;
  const int mo    = (lane & 16) ? 8 : 0;   // C/D: lanes 16-31 hold rows M=8..15
  const int kb    = (lane & 16) ? 8 : 0;   // A: lanes 16-31 start at K=8 / K=24
  constexpr int KC = FI / 32;
  constexpr int NT = FO / 16;              // even for all layers

  float sum[8], sq[8];
#pragma unroll
  for (int v = 0; v < 8; ++v) { sum[v] = 0.f; sq[v] = 0.f; }

  for (int np = 0; np < NT / 2; ++np) {
    v8f acc0 = {};
    v8f acc1 = {};
    const _Float16* wp0 = Wp + (size_t)(2 * np) * KC * 512 + lane * 16;
    const _Float16* wp1 = wp0 + (size_t)KC * 512;
#pragma unroll 4
    for (int kc = 0; kc < KC; ++kc) {
      v16h a  = load_a(inBuf, mbase, nloc, kb, kc);
      v16h b0 = *(const v16h*)wp0; wp0 += 512;
      v16h b1 = *(const v16h*)wp1; wp1 += 512;
      acc0 = __builtin_amdgcn_wmma_f32_16x16x32_f16(
          false, a, false, b0, (short)0, acc0, false, false);
      acc1 = __builtin_amdgcn_wmma_f32_16x16x32_f16(
          false, a, false, b1, (short)0, acc1, false, false);
    }
    const int n0 = (2 * np) * 16 + nloc;
    const int n1 = n0 + 16;
    const float bn0 = bias[n0];
    const float bn1 = bias[n1];
#pragma unroll
    for (int v = 0; v < 8; ++v) {
      float x0 = acc0[v] + bn0;
      float x1 = acc1[v] + bn1;
      sum[v] += x0 + x1;
      sq[v]  += x0 * x0 + x1 * x1;
      outBuf[(mbase + mo + v) * STRH + n0] = (_Float16)x0;
      outBuf[(mbase + mo + v) * STRH + n1] = (_Float16)x1;
    }
  }

  // reduce row sums across the 16 lanes of each half-wave
#pragma unroll
  for (int v = 0; v < 8; ++v) {
#pragma unroll
    for (int off = 1; off < 16; off <<= 1) {
      sum[v] += __shfl_xor(sum[v], off, 32);
      sq[v]  += __shfl_xor(sq[v],  off, 32);
    }
  }
  if ((lane & 15) == 0) {
#pragma unroll
    for (int v = 0; v < 8; ++v) {
      int r = mbase + mo + v;
      float mean = sum[v] * (1.0f / FO);
      float var  = sq[v] * (1.0f / FO) - mean * mean;
      stats[2 * r]     = mean;
      stats[2 * r + 1] = rsqrtf(var + 1e-5f);
    }
  }
  __syncthreads();

  // LayerNorm + SiLU, in place, block-wide
  const int total = BM * FO;
  for (int idx = tid; idx < total; idx += NTHREADS) {
    int r = idx / FO;          // FO is a power of two -> shift
    int c = idx & (FO - 1);
    float x  = (float)outBuf[r * STRH + c];
    float y  = (x - stats[2 * r]) * stats[2 * r + 1] * g[c] + be[c];
    y = y / (1.0f + __expf(-y));
    outBuf[r * STRH + c] = (_Float16)y;
  }
  __syncthreads();
}

// ---------------------------------------------------------------------------
// Fused C51 forward + categorical projection, one block per 128 batch rows.
// ---------------------------------------------------------------------------
__global__ __launch_bounds__(NTHREADS) void c51_fused(
    const float* __restrict__ obs, const float* __restrict__ act,
    const float* __restrict__ rew, const float* __restrict__ boot,
    const float* __restrict__ disc, const float* __restrict__ support,
    const _Float16* __restrict__ W1p, const float* __restrict__ b1,
    const float* __restrict__ g1, const float* __restrict__ be1,
    const _Float16* __restrict__ W2p, const float* __restrict__ b2,
    const float* __restrict__ g2, const float* __restrict__ be2,
    const _Float16* __restrict__ W3p, const float* __restrict__ b3,
    const float* __restrict__ g3, const float* __restrict__ be3,
    const _Float16* __restrict__ W4p, const float* __restrict__ b4,
    float* __restrict__ out)
{
  extern __shared__ char smem[];
  _Float16* bufX  = (_Float16*)smem;                                   // 135168 B
  _Float16* bufY  = (_Float16*)(smem + (size_t)BM * STRH * 2);         // 135168 B
  float*    stats = (float*)(smem + 2 * (size_t)BM * STRH * 2);        // 1024 B
  float*    logits = (float*)(smem + (size_t)BM * STRH * 2);           // overlays bufY
  float*    proj   = (float*)smem;                                     // overlays bufX
  float*    supLds = stats;                                            // reused at end

  const int tid = threadIdx.x;
  const size_t rowg0 = (size_t)blockIdx.x * BM;

  // ---- stage input: x = concat(obs, act) -> f16 into bufY -----------------
  for (int idx = tid; idx < BM * 160; idx += NTHREADS) {
    int r = idx / 160;
    int c = idx - r * 160;
    size_t gr = rowg0 + r;
    float v = (c < 128) ? obs[gr * 128 + c] : act[gr * 32 + (c - 128)];
    bufY[r * STRH + c] = (_Float16)v;
  }
  __syncthreads();

  // ---- hidden layers ------------------------------------------------------
  layer_fwd<160, 512>(bufY, bufX, W1p, b1, g1, be1, stats, tid);
  layer_fwd<512, 256>(bufX, bufY, W2p, b2, g2, be2, stats, tid);
  layer_fwd<256, 128>(bufY, bufX, W3p, b3, g3, be3, stats, tid);

  // ---- layer 4: logits (no LN), f32 into LDS, 2-way N blocking ------------
  {
    const int lane  = tid & 31;
    const int wave  = tid >> 5;
    const int mbase = wave * 16;
    const int nloc  = lane & 15;
    const int mo    = (lane & 16) ? 8 : 0;
    const int kb    = (lane & 16) ? 8 : 0;
    for (int np = 0; np < 8; ++np) {
      v8f acc0 = {};
      v8f acc1 = {};
      const _Float16* wp0 = W4p + (size_t)(2 * np) * 4 * 512 + lane * 16;
      const _Float16* wp1 = wp0 + 4 * 512;
#pragma unroll
      for (int kc = 0; kc < 4; ++kc) {
        v16h a  = load_a(bufX, mbase, nloc, kb, kc);
        v16h b0 = *(const v16h*)wp0; wp0 += 512;
        v16h b1 = *(const v16h*)wp1; wp1 += 512;
        acc0 = __builtin_amdgcn_wmma_f32_16x16x32_f16(
            false, a, false, b0, (short)0, acc0, false, false);
        acc1 = __builtin_amdgcn_wmma_f32_16x16x32_f16(
            false, a, false, b1, (short)0, acc1, false, false);
      }
      int n0 = (2 * np) * 16 + nloc;
      int n1 = n0 + 16;
      if (n0 < NATOMS) {
        float bn = b4[n0];
#pragma unroll
        for (int v = 0; v < 8; ++v)
          logits[(mbase + mo + v) * 256 + n0] = acc0[v] + bn;
      }
      if (n1 < NATOMS) {
        float bn = b4[n1];
#pragma unroll
        for (int v = 0; v < 8; ++v)
          logits[(mbase + mo + v) * 256 + n1] = acc1[v] + bn;
      }
    }
    __syncthreads();
  }

  // ---- softmax + C51 categorical projection (one thread per row) ----------
  for (int j = tid; j < NATOMS; j += NTHREADS) supLds[j] = support[j];
  __syncthreads();

  if (tid < BM) {
    const int r = tid;
    float* lg = logits + r * 256;
    float* pr = proj + r * 256;
    float mx = -3.0e38f;
    for (int j = 0; j < NATOMS; ++j) mx = fmaxf(mx, lg[j]);
    float s = 0.f;
    for (int j = 0; j < NATOMS; ++j) { float e = __expf(lg[j] - mx); lg[j] = e; s += e; }
    const float inv = 1.f / s;
    for (int j = 0; j < 256; ++j) pr[j] = 0.f;

    const size_t gr = rowg0 + r;
    const float rz = rew[gr];
    const float bd = boot[gr] * disc[gr];
    const float dz = (V_MAX_C - V_MIN_C) / (float)(NATOMS - 1);
    for (int j = 0; j < NATOMS; ++j) {
      float tz = rz + bd * supLds[j];
      tz = fminf(fmaxf(tz, V_MIN_C), V_MAX_C);
      float b = (tz - V_MIN_C) / dz;
      int lo = (int)floorf(b);
      int up = (int)ceilf(b);
      if (lo == up) {               // integer bin: reference's edge handling
        if (lo > 0) lo -= 1;
        else        up += 1;
      }
      float p = lg[j] * inv;
      pr[lo] += p * ((float)up - b);
      pr[up] += p * (b - (float)lo);
    }
  }
  __syncthreads();

  // ---- coalesced write-out ------------------------------------------------
  for (int r = 0; r < BM; ++r) {
    size_t gr = rowg0 + r;
    for (int j = tid; j < NATOMS; j += NTHREADS)
      out[gr * NATOMS + j] = proj[r * 256 + j];
  }
}

// ---------------------------------------------------------------------------
extern "C" void kernel_launch(void* const* d_in, const int* in_sizes, int n_in,
                              void* d_out, int out_size, void* d_ws, size_t ws_size,
                              hipStream_t stream) {
  (void)in_sizes; (void)n_in; (void)out_size; (void)ws_size;
  const float* obs  = (const float*)d_in[0];
  const float* act  = (const float*)d_in[1];
  const float* rew  = (const float*)d_in[2];
  const float* boot = (const float*)d_in[3];
  const float* disc = (const float*)d_in[4];
  const float* sup  = (const float*)d_in[5];
  const float* W1 = (const float*)d_in[6];  const float* b1  = (const float*)d_in[7];
  const float* g1 = (const float*)d_in[8];  const float* be1 = (const float*)d_in[9];
  const float* W2 = (const float*)d_in[10]; const float* b2  = (const float*)d_in[11];
  const float* g2 = (const float*)d_in[12]; const float* be2 = (const float*)d_in[13];
  const float* W3 = (const float*)d_in[14]; const float* b3  = (const float*)d_in[15];
  const float* g3 = (const float*)d_in[16]; const float* be3 = (const float*)d_in[17];
  const float* W4 = (const float*)d_in[18]; const float* b4  = (const float*)d_in[19];

  _Float16* W1p = (_Float16*)d_ws;               // 160*512 halves
  _Float16* W2p = W1p + 160 * 512;               // 512*256
  _Float16* W3p = W2p + 512 * 256;               // 256*128
  _Float16* W4p = W3p + 256 * 128;               // 128*256 (251 padded to 256)

  prepack_w<<<(160 * 512 + 255) / 256, 256, 0, stream>>>(W1, W1p, 160, 512, 512);
  prepack_w<<<(512 * 256 + 255) / 256, 256, 0, stream>>>(W2, W2p, 512, 256, 256);
  prepack_w<<<(256 * 128 + 255) / 256, 256, 0, stream>>>(W3, W3p, 256, 128, 128);
  prepack_w<<<(128 * 256 + 255) / 256, 256, 0, stream>>>(W4, W4p, 128, 251, 256);

  const size_t smemBytes = 2 * (size_t)BM * STRH * 2 + 1024;   // 271360 B
  c51_fused<<<131072 / BM, NTHREADS, smemBytes, stream>>>(
      obs, act, rew, boot, disc, sup,
      W1p, b1, g1, be1, W2p, b2, g2, be2, W3p, b3, g3, be3, W4p, b4,
      (float*)d_out);
}
